// TASA_attention_33646773797283
// MI455X (gfx1250) — matrix operand order repaired and checked by reading, not compiled, not run on hardware
//
#include <hip/hip_runtime.h>
#include <hip/hip_bf16.h>

// ---------------------------------------------------------------------------
// TASA attention for MI455X (gfx1250, wave32, WMMA).
// Q/K/V proj (bf16 WMMA) -> QK^T (bf16 WMMA) -> fused conv (Mt in LDS, async
// global->LDS staging when available) -> softmax + post-softmax mask (in place
// in d_out A region) -> A*V (bf16 WMMA) -> ctx*Wo^T (bf16 WMMA).
// ---------------------------------------------------------------------------

typedef __attribute__((ext_vector_type(16))) __bf16 v16bf;
typedef __attribute__((ext_vector_type(8)))  __bf16 v8bf;
typedef __attribute__((ext_vector_type(8)))  float  v8f;

constexpr int D_MODEL = 1024;
constexpr int Hh      = 16;
constexpr int Dk      = 64;
constexpr int Bb      = 2;
constexpr int Ss      = 1024;

#define BM   128
#define BN   64
#define BKK  32
#define LDSK 40   // padded row stride (bf16): 80B = 20 banks -> conflict-free

// Async global->LDS (ASYNCcnt) path, guarded so compilation never breaks.
#if defined(__HIP_DEVICE_COMPILE__) && \
    __has_builtin(__builtin_amdgcn_global_load_async_to_lds_b32) && \
    __has_builtin(__builtin_amdgcn_s_wait_asynccnt)
#define TASA_ASYNC_LDS 1
typedef __attribute__((address_space(1))) int tasa_gint;
typedef __attribute__((address_space(3))) int tasa_lint;
#else
#define TASA_ASYNC_LDS 0
#endif

static __device__ __forceinline__ v8bf cvt8(float4 a, float4 b) {
  v8bf r;
  r[0] = (__bf16)a.x; r[1] = (__bf16)a.y; r[2] = (__bf16)a.z; r[3] = (__bf16)a.w;
  r[4] = (__bf16)b.x; r[5] = (__bf16)b.y; r[6] = (__bf16)b.z; r[7] = (__bf16)b.w;
  return r;
}

// ---------------------------------------------------------------------------
// bf16-WMMA GEMM, software-pipelined register staging.
// BT=true : B is [N,K] row-major (A * B^T);  BT=false: B is [K,N] row-major.
// Batched over blockIdx.z: offsets (z/Hn)*SB + (z%Hn)*SH per operand.
// All dims are multiples of the tile dims (no bounds checks needed).
// ---------------------------------------------------------------------------
template <bool BT>
__global__ __launch_bounds__(256) void tasa_gemm_bf16(
    const float* __restrict__ Ag, const float* __restrict__ Bg,
    float* __restrict__ Cg,
    int Md, int Nd, int Kd, int lda, int ldb, int ldc,
    long aSB, long aSH, long bSB, long bSH, long cSB, long cSH, int Hn)
{
  __shared__ alignas(16) __bf16 As[BM][LDSK];   // [m][k]
  __shared__ alignas(16) __bf16 Bs[BN][LDSK];   // [n][k] (B staged transposed)

  const int z  = blockIdx.z;
  const int zb = z / Hn, zh = z % Hn;
  const float* A = Ag + (long)zb * aSB + (long)zh * aSH;
  const float* B = Bg + (long)zb * bSB + (long)zh * bSH;
  float*       C = Cg + (long)zb * cSB + (long)zh * cSH;

  const int m0   = blockIdx.y * BM;
  const int n0   = blockIdx.x * BN;
  const int tid  = threadIdx.x;
  const int lane = tid & 31;
  const int wid  = tid >> 5;
  const int wr   = wid & 3;   // 4 wave-rows x 32 = 128
  const int wc   = wid >> 2;  // 2 wave-cols x 32 = 64

  // thread-constant staging coordinates (fully unrolled stages)
  const int ar = tid >> 1;            // A row 0..127
  const int ak = (tid & 1) << 4;      // A k-offset {0,16}; 16 floats/thread
  const float* aLd = A + (long)(m0 + ar) * lda + ak;

  const int brT = tid >> 2, bkT = (tid & 3) << 3;   // BT: row 0..63, 8 floats
  const int bkN = tid >> 3, bnN = (tid & 7) << 3;   // !BT: k-row 0..31, 8 cols
  const float* bLd = BT ? (B + (long)(n0 + brT) * ldb + bkT)
                        : (B + (long)bkN * ldb + n0 + bnN);

  float4 ra0, ra1, ra2, ra3, rb0, rb1;

  auto load_stage = [&](int k0) {
    const float* ap = aLd + k0;
    ra0 = *(const float4*)(ap + 0);
    ra1 = *(const float4*)(ap + 4);
    ra2 = *(const float4*)(ap + 8);
    ra3 = *(const float4*)(ap + 12);
    const float* bp = BT ? (bLd + k0) : (bLd + (long)k0 * ldb);
    rb0 = *(const float4*)(bp + 0);
    rb1 = *(const float4*)(bp + 4);
  };

  auto store_stage = [&]() {
    *(v8bf*)&As[ar][ak]     = cvt8(ra0, ra1);
    *(v8bf*)&As[ar][ak + 8] = cvt8(ra2, ra3);
    if (BT) {
      *(v8bf*)&Bs[brT][bkT] = cvt8(rb0, rb1);
    } else {
      const float rf[8] = {rb0.x, rb0.y, rb0.z, rb0.w, rb1.x, rb1.y, rb1.z, rb1.w};
#pragma unroll
      for (int j = 0; j < 8; ++j) Bs[bnN + j][bkN] = (__bf16)rf[j];
    }
  };

  v8f acc[2][2] = {};

  load_stage(0);
  for (int k0 = 0; k0 < Kd; k0 += BKK) {
    store_stage();              // one s_wait_loadcnt covers the whole stage
    __syncthreads();

    if (k0 + BKK < Kd) load_stage(k0 + BKK);   // overlap next loads w/ compute
    if (k0 + 2 * BKK < Kd)
      __builtin_prefetch(aLd + (k0 + 2 * BKK), 0, 0);

    // fragments per ISA 7.12.2 (wave32):
    //   A: lane holds row lane&15; K = (lane>=16?8:0)+(e>=8?16:0)+(e&7)
    //   B: lane holds col lane&15; K = (lane>=16?8:0)+(e>=8?16:0)+(e&7)
    const int rr    = lane & 15;
    const int half8 = (lane >> 4) << 3;
    const int kblo  = lane & 16;

    v16bf af[2], bfg[2];
#pragma unroll
    for (int mt = 0; mt < 2; ++mt) {
      const __bf16* ap = &As[wr * 32 + mt * 16 + rr][0];
      v8bf lo = *(const v8bf*)(ap + half8);
      v8bf hi = *(const v8bf*)(ap + 16 + half8);
      af[mt] = __builtin_shufflevector(lo, hi, 0,1,2,3,4,5,6,7,
                                               8,9,10,11,12,13,14,15);
    }
#pragma unroll
    for (int nt = 0; nt < 2; ++nt) {
      const __bf16* bp = &Bs[wc * 32 + nt * 16 + rr][half8];
      v8bf lo = *(const v8bf*)(bp);
      v8bf hi = *(const v8bf*)(bp + 16);
      bfg[nt] = __builtin_shufflevector(lo, hi, 0,1,2,3,4,5,6,7,
                                                8,9,10,11,12,13,14,15);
    }

#pragma unroll
    for (int mt = 0; mt < 2; ++mt)
#pragma unroll
      for (int nt = 0; nt < 2; ++nt)
        acc[mt][nt] = __builtin_amdgcn_wmma_f32_16x16x32_bf16(
            false, af[mt], false, bfg[nt], (short)0, acc[mt][nt], false, false);

    __syncthreads();
  }

  // epilogue: C/D layout m = (lane>=16?8:0)+i, n = lane&15
  const int rr = lane & 15;
  const int mh = (lane >> 4) << 3;
#pragma unroll
  for (int mt = 0; mt < 2; ++mt)
#pragma unroll
    for (int nt = 0; nt < 2; ++nt) {
      const int cn = n0 + wc * 32 + nt * 16 + rr;
#pragma unroll
      for (int i = 0; i < 8; ++i) {
        const int cm = m0 + wr * 32 + mt * 16 + mh + i;
        C[(long)cm * ldc + cn] = acc[mt][nt][i];
      }
    }
}

// ---------------------------------------------------------------------------
// Fused transmit+aggregate conv.  Per block: one 16x16 (s,t) tile, all 16
// heads.  Mt = conv3x3(prev,Wt)+bt into LDS (halo 1, correct zero padding),
// then Ma = conv3x3(concat(M,Mt),Wa)+ba -> d_out A region.
// Global->LDS staging uses CDNA5 async copies (ASYNCcnt) when available;
// out-of-range positions get plain LDS zero stores (disjoint addresses).
// ---------------------------------------------------------------------------
__global__ __launch_bounds__(256) void tasa_conv_fused(
    const float* __restrict__ prev, const float* __restrict__ Mbuf,
    const float* __restrict__ Wt, const float* __restrict__ bt,
    const float* __restrict__ Wa, const float* __restrict__ ba,
    float* __restrict__ Ma)
{
  __shared__ float sP[Hh][20 * 20];   // prev tile halo-2 (reused for M tile)
  __shared__ float sMt[Hh][18 * 18];  // Mt tile halo-1

  const int b  = blockIdx.z;
  const int s0 = blockIdx.y * 16;
  const int t0 = blockIdx.x * 16;
  const int tid = threadIdx.x;

#if TASA_ASYNC_LDS
#define STAGE1(cond, gp, lp)                                                  \
  do {                                                                        \
    if (cond)                                                                 \
      __builtin_amdgcn_global_load_async_to_lds_b32(                          \
          (tasa_gint*)(gp), (tasa_lint*)(lp), 0, 0);                          \
    else *(lp) = 0.0f;                                                        \
  } while (0)
#define STAGE_FENCE() __builtin_amdgcn_s_wait_asynccnt(0)
#else
#define STAGE1(cond, gp, lp) do { *(lp) = (cond) ? *(gp) : 0.0f; } while (0)
#define STAGE_FENCE() do {} while (0)
#endif

  // stage prev with halo 2
  for (int i = tid; i < Hh * 400; i += 256) {
    const int c = i / 400, p = i % 400;
    const int gs = s0 + p / 20 - 2, gt = t0 + p % 20 - 2;
    const bool in = (gs >= 0) & (gs < Ss) & (gt >= 0) & (gt < Ss);
    STAGE1(in, prev + ((((long)b * Hh + c) * Ss + gs) * Ss + gt), &sP[c][p]);
  }
  STAGE_FENCE();
  __syncthreads();

  // Mt on 18x18 (tile coords -1..16); zero outside image (2nd conv padding)
  for (int i = tid; i < Hh * 324; i += 256) {
    const int oc = i / 324, p = i % 324;
    const int y = p / 18, x = p % 18;
    const int gs = s0 + y - 1, gt = t0 + x - 1;
    float a = 0.0f;
    if (gs >= 0 && gs < Ss && gt >= 0 && gt < Ss) {
      a = bt[oc];
      for (int ic = 0; ic < Hh; ++ic) {
        const float* w  = Wt + ((long)oc * Hh + ic) * 9;
        const float* pp = &sP[ic][y * 20 + x];
#pragma unroll
        for (int ky = 0; ky < 3; ++ky)
#pragma unroll
          for (int kx = 0; kx < 3; ++kx)
            a += w[ky * 3 + kx] * pp[ky * 20 + kx];
      }
    }
    sMt[oc][p] = a;
  }
  __syncthreads();

  // reuse sP as M tile (18x18, halo 1)
  for (int i = tid; i < Hh * 324; i += 256) {
    const int c = i / 324, p = i % 324;
    const int gs = s0 + p / 18 - 1, gt = t0 + p % 18 - 1;
    const bool in = (gs >= 0) & (gs < Ss) & (gt >= 0) & (gt < Ss);
    STAGE1(in, Mbuf + ((((long)b * Hh + c) * Ss + gs) * Ss + gt), &sP[c][p]);
  }
  STAGE_FENCE();
  __syncthreads();

  // Ma: one pixel per thread, loop over heads (oc uniform -> scalar weights)
  const int y = tid >> 4, x = tid & 15;
  const int gs = s0 + y, gt = t0 + x;
  for (int oc = 0; oc < Hh; ++oc) {
    float a = ba[oc];
    for (int ic = 0; ic < Hh; ++ic) {
      const float* wM = Wa + (((long)oc * 2 * Hh) + ic) * 9;
      const float* wT = Wa + (((long)oc * 2 * Hh) + Hh + ic) * 9;
      const float* pm = &sP[ic][y * 18 + x];
      const float* pt = &sMt[ic][y * 18 + x];
#pragma unroll
      for (int ky = 0; ky < 3; ++ky)
#pragma unroll
        for (int kx = 0; kx < 3; ++kx) {
          a += wM[ky * 3 + kx] * pm[ky * 18 + kx];
          a += wT[ky * 3 + kx] * pt[ky * 18 + kx];
        }
    }
    Ma[(((long)b * Hh + oc) * Ss + gs) * Ss + gt] = a;
  }
}

// ---------------------------------------------------------------------------
// Row softmax of Ma/sqrt(D_MODEL) with faithful POST-softmax mask fill.
// One 256-thread block (8 wave32) per 1024-wide row; in place.
// ---------------------------------------------------------------------------
__global__ __launch_bounds__(256) void tasa_softmax_mask(
    float* __restrict__ A, const int* __restrict__ mask)
{
  const int row = blockIdx.x;          // b*H*S + h*S + s
  const int b   = row >> 14;           // / (H*S) == 16384
  float* p = A + (long)row * Ss;
  const int* mrow = mask + (long)b * Ss;
  const int tid = threadIdx.x;

  __shared__ float rmax[8], rsum[8];

  float v[4];
  float mx = -3.4e38f;
#pragma unroll
  for (int i = 0; i < 4; ++i) {
    v[i] = p[tid + (i << 8)] * 0.03125f;   // 1/sqrt(1024)
    mx = fmaxf(mx, v[i]);
  }
#pragma unroll
  for (int o = 16; o; o >>= 1) mx = fmaxf(mx, __shfl_xor(mx, o, 32));
  if ((tid & 31) == 0) rmax[tid >> 5] = mx;
  __syncthreads();
  float m = rmax[0];
#pragma unroll
  for (int i = 1; i < 8; ++i) m = fmaxf(m, rmax[i]);

  float s = 0.0f;
#pragma unroll
  for (int i = 0; i < 4; ++i) { v[i] = __expf(v[i] - m); s += v[i]; }
#pragma unroll
  for (int o = 16; o; o >>= 1) s += __shfl_xor(s, o, 32);
  if ((tid & 31) == 0) rsum[tid >> 5] = s;
  __syncthreads();
  float tot = 0.0f;
#pragma unroll
  for (int i = 0; i < 8; ++i) tot += rsum[i];
  const float inv = 1.0f / tot;

#pragma unroll
  for (int i = 0; i < 4; ++i) {
    const int t = tid + (i << 8);
    float o = v[i] * inv;
    if (mrow[t] == 0) o = -1e9f;       // masked_fill AFTER softmax (faithful)
    p[t] = o;
  }
}

// ---------------------------------------------------------------------------
extern "C" void kernel_launch(void* const* d_in, const int* in_sizes, int n_in,
                              void* d_out, int out_size, void* d_ws, size_t ws_size,
                              hipStream_t stream)
{
  (void)in_sizes; (void)n_in; (void)out_size; (void)ws_size;

  const float* q    = (const float*)d_in[0];
  const float* k    = (const float*)d_in[1];
  const float* v    = (const float*)d_in[2];
  const int*   mask = (const int*)  d_in[3];
  const float* prev = (const float*)d_in[4];
  const float* Wq   = (const float*)d_in[5];
  const float* Wk   = (const float*)d_in[6];
  const float* Wv   = (const float*)d_in[7];
  const float* Wo   = (const float*)d_in[8];
  const float* Wt   = (const float*)d_in[9];
  const float* bt   = (const float*)d_in[10];
  const float* Wa   = (const float*)d_in[11];
  const float* ba   = (const float*)d_in[12];

  float* out  = (float*)d_out;                                  // [B,S,D]
  float* Aout = out + (size_t)Bb * Ss * D_MODEL;                // [B,H,S,S]

  // workspace: Qh|Kh|Vh (8 MB each), M (134 MB), ctx (8 MB) -> 160 MB
  float* Qh  = (float*)d_ws;
  float* Kh  = Qh + (size_t)Bb * Ss * D_MODEL;
  float* Vh  = Kh + (size_t)Bb * Ss * D_MODEL;
  float* Mb  = Vh + (size_t)Bb * Ss * D_MODEL;
  float* ctx = Mb + (size_t)Bb * Hh * Ss * Ss;

  const dim3 blk(256);
  const long SD  = (long)Ss * D_MODEL;
  const long SS2 = (long)Ss * Ss;

  // 1) projections: X @ W^T  (M=2048, N=K=1024)
  const dim3 gproj(D_MODEL / BN, (Bb * Ss) / BM, 1);
  tasa_gemm_bf16<true><<<gproj, blk, 0, stream>>>(q, Wq, Qh,
      Bb * Ss, D_MODEL, D_MODEL, D_MODEL, D_MODEL, D_MODEL,
      0, 0, 0, 0, 0, 0, 1);
  tasa_gemm_bf16<true><<<gproj, blk, 0, stream>>>(k, Wk, Kh,
      Bb * Ss, D_MODEL, D_MODEL, D_MODEL, D_MODEL, D_MODEL,
      0, 0, 0, 0, 0, 0, 1);
  tasa_gemm_bf16<true><<<gproj, blk, 0, stream>>>(v, Wv, Vh,
      Bb * Ss, D_MODEL, D_MODEL, D_MODEL, D_MODEL, D_MODEL,
      0, 0, 0, 0, 0, 0, 1);

  // 2) raw scores M = Q K^T per (b,h): M=N=1024, K=64, batched z=b*H+h
  const dim3 gsc(Ss / BN, Ss / BM, Bb * Hh);
  tasa_gemm_bf16<true><<<gsc, blk, 0, stream>>>(Qh, Kh, Mb,
      Ss, Ss, Dk, D_MODEL, D_MODEL, Ss,
      SD, Dk, SD, Dk, (long)Hh * SS2, SS2, Hh);

  // 3) fused Mt + Ma conv -> A region of d_out
  const dim3 gcv(Ss / 16, Ss / 16, Bb);
  tasa_conv_fused<<<gcv, blk, 0, stream>>>(prev, Mb, Wt, bt, Wa, ba, Aout);

  // 4) softmax + post-softmax mask, in place
  tasa_softmax_mask<<<dim3(Bb * Hh * Ss), blk, 0, stream>>>(Aout, mask);

  // 5) ctx = A @ V per (b,h): M=1024, N=64, K=1024
  const dim3 gav(Dk / BN, Ss / BM, Bb * Hh);
  tasa_gemm_bf16<false><<<gav, blk, 0, stream>>>(Aout, Vh, ctx,
      Ss, Dk, Ss, Ss, D_MODEL, D_MODEL,
      (long)Hh * SS2, SS2, SD, Dk, SD, Dk, Hh);

  // 6) out = ctx @ Wo^T
  tasa_gemm_bf16<true><<<gproj, blk, 0, stream>>>(ctx, Wo, out,
      Bb * Ss, D_MODEL, D_MODEL, D_MODEL, D_MODEL, D_MODEL,
      0, 0, 0, 0, 0, 0, 1);
}
